// GPT_6004364280079
// MI455X (gfx1250) — compile-verified, matrix operand order checked by reading, not run on hardware
//
#include <hip/hip_runtime.h>

// ---------------------------------------------------------------------------
// GPT-forward for MI455X (gfx1250): bf16 WMMA GEMMs + flash attention,
// with CDNA5 async global->LDS staging (ASYNCcnt) for the pure-copy paths.
//   B=2, T=1024, D=1024, H=16, HD=64, L=6, FF=4096, V=32000
// ---------------------------------------------------------------------------

#define TT    1024
#define DD    1024
#define HH    16
#define HDIM  64
#define LAYERS 6
#define VOCAB 32000
#define FFD   4096
#define BBATCH 2
#define MM    (BBATCH * TT)

typedef __attribute__((ext_vector_type(16))) __bf16       v16bf;
typedef __attribute__((ext_vector_type(8)))  float        v8f;
typedef __attribute__((ext_vector_type(4)))  float        f32x4;
typedef __attribute__((ext_vector_type(4)))  unsigned int u32x4;
typedef __attribute__((ext_vector_type(2)))  unsigned int u32x2;

union Frag {
    u32x4 u[2];   // two 16B groups = elements 0..7 / 8..15 of the bf16 frag
    v16bf v;
};

__device__ __forceinline__ unsigned short f2bf(float f) {
    union { float f; unsigned int u; } c; c.f = f;
    unsigned int u = c.u;
    u += 0x7fffu + ((u >> 16) & 1u);         // round-to-nearest-even
    return (unsigned short)(u >> 16);
}

__device__ __forceinline__ v8f wmma_bf16(const Frag& a, const Frag& b, v8f c) {
    return __builtin_amdgcn_wmma_f32_16x16x32_bf16(false, a.v, false, b.v,
                                                   (short)0, c, false, false);
}

// LDS byte offset of a __shared__ object (generic ptr low 32 bits = DS offset)
__device__ __forceinline__ unsigned lds_off(const void* p) {
    return (unsigned)(size_t)p;
}

// async global->LDS copies (CDNA5 TDM-adjacent path, tracked by ASYNCcnt);
// offset: immediate applies to BOTH the LDS and global side.
__device__ __forceinline__ void async_copy_b128(unsigned lds, const void* gaddr) {
    asm volatile("global_load_async_to_lds_b128 %0, %1, off"
                 :: "v"(lds), "v"(gaddr) : "memory");
}
__device__ __forceinline__ void async_copy_b128_o16(unsigned lds, const void* gaddr) {
    asm volatile("global_load_async_to_lds_b128 %0, %1, off offset:16"
                 :: "v"(lds), "v"(gaddr) : "memory");
}
__device__ __forceinline__ void async_copy_b32(unsigned lds, const void* gaddr) {
    asm volatile("global_load_async_to_lds_b32 %0, %1, off"
                 :: "v"(lds), "v"(gaddr) : "memory");
}
__device__ __forceinline__ void wait_async0() {
    asm volatile("s_wait_asynccnt 0" ::: "memory");
}
__device__ __forceinline__ void wait_ds0() {
    asm volatile("s_wait_dscnt 0" ::: "memory");
}

// max across the 16 lanes of each half-wave via ds_swizzle (immediate XOR,
// and_mask=0x1f keeps it a pure xor pattern; xor<16 never crosses halves)
__device__ __forceinline__ float half16_max(float v) {
    v = fmaxf(v, __int_as_float(__builtin_amdgcn_ds_swizzle(__float_as_int(v), 0x041f)));
    v = fmaxf(v, __int_as_float(__builtin_amdgcn_ds_swizzle(__float_as_int(v), 0x081f)));
    v = fmaxf(v, __int_as_float(__builtin_amdgcn_ds_swizzle(__float_as_int(v), 0x101f)));
    v = fmaxf(v, __int_as_float(__builtin_amdgcn_ds_swizzle(__float_as_int(v), 0x201f)));
    return v;
}

// ---------------------------------------------------------------------------
// Embedding: x[b,t,:] = tok_emb[idx[b,t],:] + pos_emb[t,:]
// ---------------------------------------------------------------------------
__global__ void embed_kernel(const int* __restrict__ idx,
                             const float* __restrict__ tok,
                             const float* __restrict__ pos,
                             float* __restrict__ x) {
    int i   = blockIdx.x * blockDim.x + threadIdx.x;   // one float4 each
    int row = i >> 8;                                  // 256 float4 per row
    int c4  = i & 255;
    int t   = row & (TT - 1);
    int tk  = idx[row];
    f32x4 a = *(const f32x4*)(tok + (size_t)tk * DD + c4 * 4);
    f32x4 p = *(const f32x4*)(pos + (size_t)t  * DD + c4 * 4);
    *(f32x4*)(x + (size_t)row * DD + c4 * 4) = a + p;
}

// ---------------------------------------------------------------------------
// LayerNorm: one 256-thread block per row (D=1024), bf16 output
// ---------------------------------------------------------------------------
__global__ void ln_kernel(const float* __restrict__ x,
                          const float* __restrict__ g,
                          const float* __restrict__ be,
                          unsigned short* __restrict__ out) {
    int row = blockIdx.x;
    int t   = threadIdx.x;                 // 256 threads, 4 floats each
    const float* xr = x + (size_t)row * DD;
    f32x4 v = *(const f32x4*)(xr + t * 4);
    float s = v.x + v.y + v.z + v.w;
    float q = v.x * v.x + v.y * v.y + v.z * v.z + v.w * v.w;
#pragma unroll
    for (int d = 1; d < 32; d <<= 1) {
        s += __shfl_xor(s, d, 32);
        q += __shfl_xor(q, d, 32);
    }
    __shared__ float ss[8], sq[8];
    int lane = t & 31, wid = t >> 5;
    if (lane == 0) { ss[wid] = s; sq[wid] = q; }
    __syncthreads();
    float ts = 0.f, tq = 0.f;
#pragma unroll
    for (int i = 0; i < 8; ++i) { ts += ss[i]; tq += sq[i]; }
    float mean = ts * (1.0f / DD);
    float var  = tq * (1.0f / DD) - mean * mean;
    float inv  = rsqrtf(var + 1e-5f);
    f32x4 gg = *(const f32x4*)(g  + t * 4);
    f32x4 bb = *(const f32x4*)(be + t * 4);
    unsigned short o0 = f2bf((v.x - mean) * inv * gg.x + bb.x);
    unsigned short o1 = f2bf((v.y - mean) * inv * gg.y + bb.y);
    unsigned short o2 = f2bf((v.z - mean) * inv * gg.z + bb.z);
    unsigned short o3 = f2bf((v.w - mean) * inv * gg.w + bb.w);
    u32x2 wv;
    wv.x = (unsigned)o0 | ((unsigned)o1 << 16);
    wv.y = (unsigned)o2 | ((unsigned)o3 << 16);
    *(u32x2*)(out + (size_t)row * DD + t * 4) = wv;
}

// ---------------------------------------------------------------------------
// Generic bf16 WMMA GEMM:  out[M,N] = A[M,K](bf16) * W[K,N](f32->bf16)
//   BM=128 BN=128 BK=32, 256 threads = 8 waves (4 row-groups x 2 col-groups),
//   each wave computes a 32x64 tile = 2x4 WMMA 16x16 tiles.
//   A tile staged via async global->LDS (pure bf16 copy, no VGPR round trip).
// HEADED: W element (k,n) at W + (n>>6)*headStride + k*ldw + (n&63)
//   (for the (H,D,HD) Wq/Wk/Wv layout; ldw = HD = 64).
// ---------------------------------------------------------------------------
template<bool OB16, bool RELU, bool BIAS, bool RES, bool HEADED>
__global__ __launch_bounds__(256) void gemm_kernel(
    const unsigned short* __restrict__ A, const float* __restrict__ W,
    const float* __restrict__ bias, const float* __restrict__ res,
    void* __restrict__ outv, int M, int N, int K, int ldw, long headStride)
{
    __shared__ __align__(16) unsigned short sA[128 * 40];   // [m][k], pad 40
    __shared__ __align__(16) unsigned short sB[128 * 40];   // [n][k], pad 40

    const int tid  = threadIdx.x;
    const int lane = tid & 31;
    const int wid  = tid >> 5;
    const int ln   = lane & 15;
    const int sel  = lane >> 4;        // half-wave select (K-group per layout)
    const int m0   = blockIdx.y * 128;
    const int n0   = blockIdx.x * 128;
    const int wrow = (wid & 3) * 32;
    const int wcol = (wid >> 2) * 64;

    v8f acc[2][4] = {};

    const int ar = tid >> 1;           // A stage: row, 16-elem chunk
    const int ac = (tid & 1) * 16;
    const int bk = tid >> 3;           // B stage: k row, 16-col chunk
    const int bn = (tid & 7) * 16;

    const unsigned laddrA = lds_off(&sA[ar * 40 + ac]);

    for (int k0 = 0; k0 < K; k0 += 32) {
        // ---- stage A tile: async global->LDS b128 x2 (32B per thread) ----
        {
            const unsigned short* ga = A + (size_t)(m0 + ar) * K + k0 + ac;
            async_copy_b128(laddrA, ga);
            async_copy_b128_o16(laddrA, ga);
        }
        // ---- stage B tile (f32 load, bf16 convert, transposed store) ----
        {
            const float* wp;
            if (HEADED) {
                int head = (n0 + bn) >> 6;
                wp = W + (size_t)head * headStride + (size_t)(k0 + bk) * ldw
                       + ((n0 + bn) & 63);
            } else {
                wp = W + (size_t)(k0 + bk) * ldw + (n0 + bn);
            }
            union { f32x4 v[4]; float f[16]; } wu;
            wu.v[0] = *(const f32x4*)(wp);
            wu.v[1] = *(const f32x4*)(wp + 4);
            wu.v[2] = *(const f32x4*)(wp + 8);
            wu.v[3] = *(const f32x4*)(wp + 12);
#pragma unroll
            for (int i = 0; i < 16; ++i)
                sB[(bn + i) * 40 + bk] = f2bf(wu.f[i]);
        }
        wait_async0();                 // A-tile async writes landed in LDS
        __syncthreads();

        // ---- fragments + 8 WMMAs ----
        Frag af[2], bfr[4];
#pragma unroll
        for (int i = 0; i < 2; ++i) {
            const unsigned short* p = &sA[(wrow + i * 16 + ln) * 40];
            af[i].u[0] = *(const u32x4*)(p + sel * 8);
            af[i].u[1] = *(const u32x4*)(p + 16 + sel * 8);
        }
#pragma unroll
        for (int j = 0; j < 4; ++j) {
            const unsigned short* p = &sB[(wcol + j * 16 + ln) * 40];
            bfr[j].u[0] = *(const u32x4*)(p + sel * 8);
            bfr[j].u[1] = *(const u32x4*)(p + 16 + sel * 8);
        }
#pragma unroll
        for (int i = 0; i < 2; ++i)
#pragma unroll
            for (int j = 0; j < 4; ++j)
                acc[i][j] = wmma_bf16(af[i], bfr[j], acc[i][j]);
        __syncthreads();
    }

    // ---- epilogue (C layout: VGPR r -> row r (lanes 0-15) / r+8 (16-31)) ----
#pragma unroll
    for (int i = 0; i < 2; ++i)
#pragma unroll
        for (int j = 0; j < 4; ++j)
#pragma unroll
            for (int r = 0; r < 8; ++r) {
                size_t row = (size_t)(m0 + wrow + i * 16 + sel * 8 + r);
                size_t col = (size_t)(n0 + wcol + j * 16 + ln);
                float v = acc[i][j][r];
                if constexpr (BIAS) v += bias[col];
                if constexpr (RES)  v += res[row * (size_t)N + col];
                if constexpr (RELU) v = fmaxf(v, 0.f);
                if constexpr (OB16)
                    ((unsigned short*)outv)[row * (size_t)N + col] = f2bf(v);
                else
                    ((float*)outv)[row * (size_t)N + col] = v;
            }
}

// ---------------------------------------------------------------------------
// Flash attention: one wave per 16-row query tile, 32-key tiles.
//   S = Q K^T (4 WMMAs), online softmax in C-layout regs (max via ds_swizzle,
//   row-sums via an extra WMMA against an all-ones B fragment), P via LDS to
//   A-layout, O += P V (4 WMMAs). K tile staged via async global->LDS.
//   q/k/v/out all bf16 [B,T,H,HD]-as-[M,D].
// ---------------------------------------------------------------------------
__global__ __launch_bounds__(128) void attn_kernel(
    const unsigned short* __restrict__ qb,
    const unsigned short* __restrict__ kb,
    const unsigned short* __restrict__ vb,
    unsigned short* __restrict__ ob)
{
    __shared__ __align__(16) unsigned short sK [4][32 * 72];  // [key][hd] pad 72
    __shared__ __align__(16) unsigned short sVt[4][64 * 40];  // [hd][key] pad 40
    __shared__ __align__(16) unsigned short sP [4][16 * 40];  // [qrow][key] pad 40

    const int lane = threadIdx.x & 31;
    const int wid  = threadIdx.x >> 5;
    const int ln   = lane & 15;
    const int sel  = lane >> 4;
    const int b    = blockIdx.y >> 4;
    const int h    = blockIdx.y & 15;
    const int row0 = (blockIdx.x * 4 + wid) * 16;

    // Q fragments (A layout), direct from global bf16
    const unsigned short* qp = qb + (size_t)(b * TT + row0) * DD + h * HDIM;
    Frag aq[2];
#pragma unroll
    for (int c = 0; c < 2; ++c) {
        const unsigned short* p = qp + (size_t)ln * DD + c * 32;
        aq[c].u[0] = *(const u32x4*)(p + sel * 8);
        aq[c].u[1] = *(const u32x4*)(p + 16 + sel * 8);
    }

    // all-ones bf16 B fragment: one extra WMMA yields per-row sums of P
    Frag onesf;
    onesf.u[0] = (u32x4)(0x3f803f80u);
    onesf.u[1] = (u32x4)(0x3f803f80u);

    float rm[8];
    v8f o[4] = {};
    v8f osum = {};                        // row-sum accumulator (all cols equal)
#pragma unroll
    for (int r = 0; r < 8; ++r) rm[r] = -3.0e38f;

    const unsigned laddrK = lds_off(&sK[wid][2 * lane]);

    const int ntiles = (row0 + 47) >> 5;   // causal: keys up to row0+15
    for (int jt = 0; jt < ntiles; ++jt) {
        const int k0 = jt * 32;
        const unsigned short* kp = kb + (size_t)(b * TT + k0) * DD + h * HDIM;
        const unsigned short* vp = vb + (size_t)(b * TT + k0) * DD + h * HDIM;
        // stage K rows via async global->LDS b32; V transposed via VGPRs
#pragma unroll 8
        for (int r = 0; r < 32; ++r) {
            async_copy_b32(laddrK + r * 72 * 2, kp + (size_t)r * DD + 2 * lane);
            unsigned int vv = *(const unsigned int*)(vp + (size_t)r * DD + 2 * lane);
            sVt[wid][(2 * lane)     * 40 + r] = (unsigned short)(vv & 0xffffu);
            sVt[wid][(2 * lane + 1) * 40 + r] = (unsigned short)(vv >> 16);
        }
        wait_async0();                 // K async writes visible
        wait_ds0();                    // V transpose stores visible

        // S = Q K^T : two 16-key subtiles, K-dim 64 = 2 WMMAs each
        v8f s[2];
#pragma unroll
        for (int sub = 0; sub < 2; ++sub) {
            const unsigned short* kr = &sK[wid][(sub * 16 + ln) * 72];
            Frag kf0, kf1;
            kf0.u[0] = *(const u32x4*)(kr + sel * 8);
            kf0.u[1] = *(const u32x4*)(kr + 16 + sel * 8);
            kf1.u[0] = *(const u32x4*)(kr + 32 + sel * 8);
            kf1.u[1] = *(const u32x4*)(kr + 48 + sel * 8);
            v8f a = {};
            a = wmma_bf16(aq[0], kf0, a);
            a = wmma_bf16(aq[1], kf1, a);
            s[sub] = a;
        }

        // online softmax per row (rows live across 16-lane halves)
#pragma unroll
        for (int r = 0; r < 8; ++r) {
            int grow = row0 + sel * 8 + r;
            float v0 = s[0][r] * 0.125f;          // 1/sqrt(64)
            float v1 = s[1][r] * 0.125f;
            if (k0 + ln > grow)      v0 = -3.0e38f;
            if (k0 + 16 + ln > grow) v1 = -3.0e38f;
            float mx   = half16_max(fmaxf(v0, v1));
            float mnew = fmaxf(rm[r], mx);
            float al   = __expf(rm[r] - mnew);
            rm[r] = mnew;
            float p0 = __expf(v0 - mnew);
            float p1 = __expf(v1 - mnew);
            osum[r] *= al;
#pragma unroll
            for (int c = 0; c < 4; ++c) o[c][r] *= al;
            sP[wid][(sel * 8 + r) * 40 + ln]      = f2bf(p0);
            sP[wid][(sel * 8 + r) * 40 + 16 + ln] = f2bf(p1);
        }
        wait_ds0();

        // O += P V : P in A layout from LDS, V^T rows as B fragments;
        // row-sum accumulator uses the all-ones B fragment.
        Frag pf;
        const unsigned short* pr = &sP[wid][ln * 40];
        pf.u[0] = *(const u32x4*)(pr + sel * 8);
        pf.u[1] = *(const u32x4*)(pr + 16 + sel * 8);
        osum = wmma_bf16(pf, onesf, osum);
#pragma unroll
        for (int c = 0; c < 4; ++c) {
            const unsigned short* vr = &sVt[wid][(c * 16 + ln) * 40];
            Frag vf;
            vf.u[0] = *(const u32x4*)(vr + sel * 8);
            vf.u[1] = *(const u32x4*)(vr + 16 + sel * 8);
            o[c] = wmma_bf16(pf, vf, o[c]);
        }
    }

    // normalize and write bf16 heads-concat layout [B,T,H*HD]
#pragma unroll
    for (int c = 0; c < 4; ++c)
#pragma unroll
        for (int r = 0; r < 8; ++r) {
            int grow = row0 + sel * 8 + r;
            float val = o[c][r] / osum[r];
            ob[(size_t)(b * TT + grow) * DD + h * HDIM + c * 16 + ln] = f2bf(val);
        }
}

// ---------------------------------------------------------------------------
// Orchestration
// ---------------------------------------------------------------------------
extern "C" void kernel_launch(void* const* d_in, const int* in_sizes, int n_in,
                              void* d_out, int out_size, void* d_ws, size_t ws_size,
                              hipStream_t stream) {
    const int*   idx  = (const int*)  d_in[0];
    const float* tok  = (const float*)d_in[1];
    const float* pos  = (const float*)d_in[2];
    const float* Wq   = (const float*)d_in[3];
    const float* Wk   = (const float*)d_in[4];
    const float* Wv   = (const float*)d_in[5];
    const float* Wo   = (const float*)d_in[6];
    const float* bo   = (const float*)d_in[7];
    const float* ln1g = (const float*)d_in[8];
    const float* ln1b = (const float*)d_in[9];
    const float* ln2g = (const float*)d_in[10];
    const float* ln2b = (const float*)d_in[11];
    const float* W1   = (const float*)d_in[12];
    const float* b1   = (const float*)d_in[13];
    const float* W2   = (const float*)d_in[14];
    const float* b2   = (const float*)d_in[15];
    const float* lnfg = (const float*)d_in[16];
    const float* lnfb = (const float*)d_in[17];
    const float* Wh   = (const float*)d_in[18];
    const float* bh   = (const float*)d_in[19];
    (void)in_sizes; (void)n_in; (void)out_size; (void)ws_size;

    char* wsp = (char*)d_ws;
    size_t off = 0;
    auto carve = [&](size_t bytes) -> void* {
        void* p = wsp + off;
        off += (bytes + 255) & ~(size_t)255;
        return p;
    };
    float*          x   = (float*)         carve((size_t)MM * DD * 4);   //  8 MB
    unsigned short* lnb = (unsigned short*)carve((size_t)MM * DD * 2);   //  4 MB
    unsigned short* qb  = (unsigned short*)carve((size_t)MM * DD * 2);
    unsigned short* kbf = (unsigned short*)carve((size_t)MM * DD * 2);
    unsigned short* vbf = (unsigned short*)carve((size_t)MM * DD * 2);
    unsigned short* ab  = (unsigned short*)carve((size_t)MM * DD * 2);
    unsigned short* fb  = (unsigned short*)carve((size_t)MM * FFD * 2);  // 16 MB

    embed_kernel<<<MM * DD / 1024, 256, 0, stream>>>(idx, tok, pos, x);

    const dim3 gD(DD / 128, MM / 128);       // 8 x 16
    const dim3 gF(FFD / 128, MM / 128);      // 32 x 16
    const dim3 gV(VOCAB / 128, MM / 128);    // 250 x 16
    const dim3 gA(TT / 64, BBATCH * HH);     // 16 x 32, 128 thr (4 waves)

    for (int l = 0; l < LAYERS; ++l) {
        ln_kernel<<<MM, 256, 0, stream>>>(x, ln1g + l * DD, ln1b + l * DD, lnb);

        const size_t wqk = (size_t)l * HH * DD * HDIM;
        gemm_kernel<true,false,false,false,true><<<gD, 256, 0, stream>>>(
            lnb, Wq + wqk, nullptr, nullptr, qb,  MM, DD, DD, HDIM, (long)DD * HDIM);
        gemm_kernel<true,false,false,false,true><<<gD, 256, 0, stream>>>(
            lnb, Wk + wqk, nullptr, nullptr, kbf, MM, DD, DD, HDIM, (long)DD * HDIM);
        gemm_kernel<true,false,false,false,true><<<gD, 256, 0, stream>>>(
            lnb, Wv + wqk, nullptr, nullptr, vbf, MM, DD, DD, HDIM, (long)DD * HDIM);

        attn_kernel<<<gA, 128, 0, stream>>>(qb, kbf, vbf, ab);

        gemm_kernel<false,false,true,true,false><<<gD, 256, 0, stream>>>(
            ab, Wo + (size_t)l * DD * DD, bo + l * DD, x, x, MM, DD, DD, DD, 0);

        ln_kernel<<<MM, 256, 0, stream>>>(x, ln2g + l * DD, ln2b + l * DD, lnb);

        gemm_kernel<true,true,true,false,false><<<gF, 256, 0, stream>>>(
            lnb, W1 + (size_t)l * DD * FFD, b1 + l * FFD, nullptr, fb,
            MM, FFD, DD, FFD, 0);
        gemm_kernel<false,false,true,true,false><<<gD, 256, 0, stream>>>(
            fb, W2 + (size_t)l * FFD * DD, b2 + l * DD, x, x, MM, DD, FFD, DD, 0);
    }

    ln_kernel<<<MM, 256, 0, stream>>>(x, lnfg, lnfb, lnb);
    gemm_kernel<false,false,true,false,false><<<gV, 256, 0, stream>>>(
        lnb, Wh, bh, nullptr, d_out, MM, VOCAB, DD, VOCAB, 0);
}
